// Model_Base_70970039599490
// MI455X (gfx1250) — compile-verified, matrix operand order checked by reading, not run on hardware
//
#include <hip/hip_runtime.h>

typedef __attribute__((ext_vector_type(16))) _Float16 v16h;
typedef __attribute__((ext_vector_type(8)))  float    v8f;

// Problem dims (fixed by the reference): B=16, N=M=2048, d=256
#define B_   16
#define NN   2048
#define DD   256
#define CH   16                        // n-chunks for stage-1 reduction
#define ROWS_PER_CH (NN / CH)          // 128
#define PARTIALS_FLOATS (2 * B_ * CH * DD)   // 131072 floats = 512 KB
#define SUMS_OFF PARTIALS_FLOATS             // then 2*B_*DD floats of sums

// ---------------------------------------------------------------------------
// Stage 1: partial column sums.  Block (which,b,chunk) reduces 128 rows of
// [row, 256] into one partial row.  Thread dd strides row-by-row: for a fixed
// row the 256 threads read 1 KB contiguously -> perfectly coalesced stream.
// ---------------------------------------------------------------------------
__global__ void sum_partial(const float* __restrict__ x,
                            const float* __restrict__ y,
                            float* __restrict__ ws) {
    const int blk   = blockIdx.x;            // [0, 2*B_*CH)
    const int dd    = threadIdx.x;           // 0..255
    const int c     = blk % CH;
    const int b     = (blk / CH) % B_;
    const int which = blk / (CH * B_);       // 0 => x, 1 => y
    const float* src = which ? y : x;
    const float* p = src + ((size_t)(b * NN + c * ROWS_PER_CH)) * DD + dd;
    float acc = 0.f;
#pragma unroll 4
    for (int i = 0; i < ROWS_PER_CH; ++i) {
        if (i + 8 < ROWS_PER_CH)
            __builtin_prefetch(p + (size_t)(i + 8) * DD, 0, 0);  // global_prefetch_b8
        acc += p[(size_t)i * DD];
    }
    ws[((which * B_ + b) * CH + c) * DD + dd] = acc;
}

// ---------------------------------------------------------------------------
// Stage 2: fold the 16 partials -> xsum[b,:], ysum[b,:] (8192 floats total).
// ---------------------------------------------------------------------------
__global__ void sum_final(float* __restrict__ ws) {
    const int t  = blockIdx.x * blockDim.x + threadIdx.x;  // [0, 2*B_*DD)
    const int dd = t % DD;
    const int bb = t / DD;                                  // which*B_ + b
    float acc = 0.f;
#pragma unroll
    for (int c = 0; c < CH; ++c) acc += ws[(bb * CH + c) * DD + dd];
    ws[SUMS_OFF + bb * DD + dd] = acc;
}

// ---------------------------------------------------------------------------
// Stage 3: GEMV via WMMA broadcast.  One wave per 16-row tile.
//   which==0: rowsum[b,m] = y[b,m,:] . xsum[b,:]
//   which==1: colsum[b,n] = x[b,n,:] . ysum[b,:]
// A (16x32 f16): lane L<16 holds row M=L, K = kb+{0..7, 16..23};
//                lane L>=16 holds row M=L-16, K = kb+{8..15, 24..31}.
// B (32x16 f16): vector broadcast into all 16 columns (same K mapping),
// so every column of D carries the same dot products.
// C/D f32 layout: lane 0 -> rows M=0..7 in c[0..7]; lane 16 -> rows M=8..15.
// ---------------------------------------------------------------------------
__global__ void gemv_wmma(const float* __restrict__ x,
                          const float* __restrict__ y,
                          const float* __restrict__ ws,
                          float* __restrict__ out) {
    const int lane = threadIdx.x & 31;
    const int wave = blockIdx.x * (blockDim.x >> 5) + (threadIdx.x >> 5);

    const int tiles_per_b = NN / 16;                 // 128
    const int tile  = wave % tiles_per_b;
    const int b     = (wave / tiles_per_b) % B_;
    const int which = wave / (tiles_per_b * B_);     // 0: rowsum, 1: colsum

    const float* mat = which ? x : y;                         // matrix operand
    const float* vec = ws + SUMS_OFF + (which * B_ + b) * DD; // xsum or ysum

    const int m  = 16 * tile + (lane & 15);
    const int kh = (lane >> 4) * 8;                  // 0 or 8 (K sub-block)
    const float* mrow = mat + ((size_t)(b * NN + m)) * DD;

    v8f c = {};
#pragma unroll
    for (int kb = 0; kb < DD; kb += 32) {
        const float4* am0 = (const float4*)(mrow + kb + kh);
        const float4* am1 = (const float4*)(mrow + kb + 16 + kh);
        const float4* bv0 = (const float4*)(vec + kb + kh);
        const float4* bv1 = (const float4*)(vec + kb + 16 + kh);
        float4 a0 = am0[0], a1 = am0[1], a2 = am1[0], a3 = am1[1];
        float4 s0 = bv0[0], s1 = bv0[1], s2 = bv1[0], s3 = bv1[1];

        v16h A, Bv;
        A[0]  = (_Float16)a0.x; A[1]  = (_Float16)a0.y;
        A[2]  = (_Float16)a0.z; A[3]  = (_Float16)a0.w;
        A[4]  = (_Float16)a1.x; A[5]  = (_Float16)a1.y;
        A[6]  = (_Float16)a1.z; A[7]  = (_Float16)a1.w;
        A[8]  = (_Float16)a2.x; A[9]  = (_Float16)a2.y;
        A[10] = (_Float16)a2.z; A[11] = (_Float16)a2.w;
        A[12] = (_Float16)a3.x; A[13] = (_Float16)a3.y;
        A[14] = (_Float16)a3.z; A[15] = (_Float16)a3.w;

        Bv[0]  = (_Float16)s0.x; Bv[1]  = (_Float16)s0.y;
        Bv[2]  = (_Float16)s0.z; Bv[3]  = (_Float16)s0.w;
        Bv[4]  = (_Float16)s1.x; Bv[5]  = (_Float16)s1.y;
        Bv[6]  = (_Float16)s1.z; Bv[7]  = (_Float16)s1.w;
        Bv[8]  = (_Float16)s2.x; Bv[9]  = (_Float16)s2.y;
        Bv[10] = (_Float16)s2.z; Bv[11] = (_Float16)s2.w;
        Bv[12] = (_Float16)s3.x; Bv[13] = (_Float16)s3.y;
        Bv[14] = (_Float16)s3.z; Bv[15] = (_Float16)s3.w;

        // D = A*B + C   (f32 accumulate)
        c = __builtin_amdgcn_wmma_f32_16x16x32_f16(
                /*neg_a=*/false, A, /*neg_b=*/false, Bv,
                /*c_mod=*/(short)0, c, /*reuse_a=*/false, /*reuse_b=*/false);
    }

    // out layout: [B, M+N]; rowsum occupies [0,2048), colsum [2048,4096)
    const int obase = b * (2 * NN) + which * NN + 16 * tile;
    if (lane == 0) {
#pragma unroll
        for (int i = 0; i < 8; ++i) out[obase + i] = c[i];
    } else if (lane == 16) {
#pragma unroll
        for (int i = 0; i < 8; ++i) out[obase + 8 + i] = c[i];
    }
}

extern "C" void kernel_launch(void* const* d_in, const int* in_sizes, int n_in,
                              void* d_out, int out_size, void* d_ws, size_t ws_size,
                              hipStream_t stream) {
    (void)in_sizes; (void)n_in; (void)out_size; (void)ws_size;
    const float* x  = (const float*)d_in[0];
    const float* y  = (const float*)d_in[1];
    float*       out = (float*)d_out;
    float*       ws  = (float*)d_ws;

    // Stage 1: 512 blocks x 256 threads, streams all 67 MB once.
    sum_partial<<<2 * B_ * CH, DD, 0, stream>>>(x, y, ws);
    // Stage 2: 8192 threads fold partials.
    sum_final<<<(2 * B_ * DD) / 256, 256, 0, stream>>>(ws);
    // Stage 3: 4096 waves (one per 16-row tile), 8 waves/block -> 512 blocks.
    gemv_wmma<<<(2 * B_ * (NN / 16)) / 8, 256, 0, stream>>>(x, y, ws, out);
}
